// proposedMethods_54760833024656
// MI455X (gfx1250) — matrix operand order repaired and checked by reading, not compiled, not run on hardware
//
#include <hip/hip_runtime.h>
#include <math.h>

typedef __attribute__((ext_vector_type(16))) _Float16     v16h;
typedef __attribute__((ext_vector_type(8)))  _Float16     v8h;
typedef __attribute__((ext_vector_type(8)))  float        v8f;
typedef __attribute__((ext_vector_type(8)))  unsigned int v8u;

constexpr int B_TOT = 2048;
constexpr int S_TOT = 9;
constexpr int M2    = 64;
constexpr int G_TOT = 181;
constexpr int L_TOT = 16;
constexpr int GP    = 192;   // G padded to 12 tiles of 16
constexpr int BT    = 32;    // batch tile per workgroup

// D = A(16x32 f16) * B(32x16 f16) + C(16x16 f32), fp32 accumulate
__device__ __forceinline__ v8f wmma16(v16h a, v16h b, v8f c) {
  return __builtin_amdgcn_wmma_f32_16x16x32_f16(
      /*neg_a=*/false, a, /*neg_b=*/false, b,
      /*c_mod=*/(short)0, c, /*reuse_a=*/false, /*reuse_b=*/false);
}

__device__ __forceinline__ v16h neg16(v16h x) {
  v8u u = __builtin_bit_cast(v8u, x);
  u ^= 0x80008000u;                 // flip both packed f16 sign bits
  return __builtin_bit_cast(v16h, u);
}

// A-fragment: lane holds row m = lane&15;
// halves 0..15 = K (k0 + sel*16 .. +15), contiguous (matches B fills)
__device__ __forceinline__ v16h load_afrag(const _Float16* rowp, int k0, int sel) {
  const _Float16* p = rowp + k0 + sel * 16;
  return *(const v16h*)(p);
}

// ---------------- Phase 1a: transposed dictionary, split f16 hi/lo ----------
__global__ void prep_dict(const float* __restrict__ dr, const float* __restrict__ di,
                          _Float16* __restrict__ drt_h, _Float16* __restrict__ drt_l,
                          _Float16* __restrict__ dit_h, _Float16* __restrict__ dit_l) {
  int i = blockIdx.x * 256 + threadIdx.x;
  if (i >= S_TOT * GP * M2) return;
  int m = i & 63;
  int g = (i >> 6) % GP;
  int s = i / (GP * M2);
  float vr = 0.0f, vi = 0.0f;
  if (g < G_TOT) {
    size_t src = ((size_t)s * M2 + m) * G_TOT + g;
    vr = dr[src]; vi = di[src];
  }
  _Float16 h = (_Float16)vr; drt_h[i] = h; drt_l[i] = (_Float16)(vr - (float)h);
  h = (_Float16)vi;          dit_h[i] = h; dit_l[i] = (_Float16)(vi - (float)h);
}

// ---------------- Phase 1b: Gram matrix DhD = D^H D, split f16 hi/lo --------
__global__ void prep_gram(const float* __restrict__ dr, const float* __restrict__ di,
                          _Float16* __restrict__ arh, _Float16* __restrict__ arl,
                          _Float16* __restrict__ aih, _Float16* __restrict__ ail) {
  int i = blockIdx.x * 256 + threadIdx.x;
  if (i >= S_TOT * GP * GP) return;
  int hc = i % GP;
  int g  = (i / GP) % GP;
  int s  = i / (GP * GP);
  float ar = 0.0f, ai = 0.0f;
  if (g < G_TOT && hc < G_TOT) {
    const float* Drs = dr + (size_t)s * M2 * G_TOT;
    const float* Dis = di + (size_t)s * M2 * G_TOT;
    for (int m = 0; m < M2; ++m) {
      float drg = Drs[m * G_TOT + g], drh = Drs[m * G_TOT + hc];
      float dig = Dis[m * G_TOT + g], dih = Dis[m * G_TOT + hc];
      ar += drg * drh + dig * dih;   // Re(conj(D[m,g]) * D[m,h])
      ai += drg * dih - dig * drh;   // Im(...)
    }
  }
  _Float16 t = (_Float16)ar; arh[i] = t; arl[i] = (_Float16)(ar - (float)t);
  t = (_Float16)ai;          aih[i] = t; ail[i] = (_Float16)(ai - (float)t);
}

// ---------------- Phase 2: Dhy + 16 unfolded layers + softmax ---------------
__global__ __launch_bounds__(384) void lista_main(
    const float* __restrict__ covr, const float* __restrict__ covi,
    const float* __restrict__ theta, const float* __restrict__ gamma,
    const _Float16* __restrict__ drt_h, const _Float16* __restrict__ drt_l,
    const _Float16* __restrict__ dit_h, const _Float16* __restrict__ dit_l,
    const _Float16* __restrict__ arh, const _Float16* __restrict__ arl,
    const _Float16* __restrict__ aih, const _Float16* __restrict__ ail,
    float* __restrict__ out)
{
  __shared__ __align__(32) _Float16 sh_yhr[BT * M2], sh_ylr[BT * M2];
  __shared__ __align__(32) _Float16 sh_yhi[BT * M2], sh_yli[BT * M2];
  __shared__ __align__(32) _Float16 sh_rh[BT * GP], sh_rl[BT * GP];
  __shared__ float sh_red[12 * 32];
  __shared__ float sh_stat[32];

  const int s    = blockIdx.y;
  const int b0   = blockIdx.x * BT;
  const int tid  = threadIdx.x;
  const int lane = tid & 31;
  const int wave = tid >> 5;          // 0..11 == g-tile index
  const int sel  = lane >> 4;         // upper/lower half of the wave
  const int col  = lane & 15;

  // ---- stage y tile (covariance vectors) into LDS as split f16 ----
  for (int i = tid; i < BT * M2; i += 384) {
    int bl = i >> 6, m = i & 63;
    size_t src = ((size_t)(b0 + bl) * S_TOT + s) * M2 + m;
    float vr = covr[src], vi = covi[src];
    _Float16 h = (_Float16)vr; sh_yhr[i] = h; sh_ylr[i] = (_Float16)(vr - (float)h);
    h = (_Float16)vi;          sh_yhi[i] = h; sh_yli[i] = (_Float16)(vi - (float)h);
  }
  __syncthreads();

  const int g0   = wave * 16;
  const int grow = g0 + col;          // A-operand row for this lane

  const _Float16* pdrh = drt_h + ((size_t)s * GP + grow) * M2;
  const _Float16* pdrl = drt_l + ((size_t)s * GP + grow) * M2;
  const _Float16* pdih = dit_h + ((size_t)s * GP + grow) * M2;
  const _Float16* pdil = dit_l + ((size_t)s * GP + grow) * M2;

  // ---- Dhy = D^H y  (K = 64 -> two K=32 WMMA tiles), kept in registers ----
  v8f dhyr[2], dhyi[2], r[2];
  #pragma unroll
  for (int bt = 0; bt < 2; ++bt) {
    v8f accr = {}; v8f acci = {};
    #pragma unroll
    for (int kt = 0; kt < 2; ++kt) {
      const int k0 = kt * 32;
      v16h aDrh = load_afrag(pdrh, k0, sel);
      v16h aDrl = load_afrag(pdrl, k0, sel);
      v16h aDih = load_afrag(pdih, k0, sel);
      v16h aDil = load_afrag(pdil, k0, sel);
      const int yoff = (bt * 16 + col) * M2 + k0 + sel * 16;
      v16h yrh = *(const v16h*)(sh_yhr + yoff);
      v16h yrl = *(const v16h*)(sh_ylr + yoff);
      v16h yih = *(const v16h*)(sh_yhi + yoff);
      v16h yil = *(const v16h*)(sh_yli + yoff);
      v16h yrnh = neg16(yrh), yrnl = neg16(yrl);
      // Re(Dhy) = Dr^T yr + Di^T yi
      accr = wmma16(aDrh, yrh, accr); accr = wmma16(aDrh, yrl, accr); accr = wmma16(aDrl, yrh, accr);
      accr = wmma16(aDih, yih, accr); accr = wmma16(aDih, yil, accr); accr = wmma16(aDil, yih, accr);
      // Im(Dhy) = Dr^T yi - Di^T yr
      acci = wmma16(aDrh, yih,  acci); acci = wmma16(aDrh, yil,  acci); acci = wmma16(aDrl, yih,  acci);
      acci = wmma16(aDih, yrnh, acci); acci = wmma16(aDih, yrnl, acci); acci = wmma16(aDil, yrnh, acci);
    }
    dhyr[bt] = accr; dhyi[bt] = acci; r[bt] = accr;   // r0 = Re(Dhy)
  }

  // ---- publish r0 to LDS as split f16 (B-operand layout: [b][g] row-major) ----
  #pragma unroll
  for (int bt = 0; bt < 2; ++bt) {
    int bl = bt * 16 + col;
    #pragma unroll
    for (int j = 0; j < 8; ++j) {
      int g = g0 + j + 8 * sel;       // C/D layout: M = j + 8*sel
      float v = r[bt][j];
      _Float16 h = (_Float16)v;
      sh_rh[bl * GP + g] = h;
      sh_rl[bl * GP + g] = (_Float16)(v - (float)h);
    }
  }
  __syncthreads();

  const _Float16* parh = arh + ((size_t)s * GP + grow) * GP;
  const _Float16* parl = arl + ((size_t)s * GP + grow) * GP;
  const _Float16* paih = aih + ((size_t)s * GP + grow) * GP;
  const _Float16* pail = ail + ((size_t)s * GP + grow) * GP;

  const size_t OUT2 = (size_t)B_TOT * S_TOT * G_TOT;   // offset of result_init_all

  #pragma unroll 1
  for (int l = 0; l < L_TOT; ++l) {
    const float th = theta[l];
    const float ga = gamma[l];
    v8f tr[2] = { {}, {} };           // Re(DhD @ r)
    v8f ti[2] = { {}, {} };           // Im(DhD @ r)
    #pragma unroll 1
    for (int kt = 0; kt < 6; ++kt) {  // K = 192 in 6 tiles of 32
      const int k0 = kt * 32;
      v16h fArh = load_afrag(parh, k0, sel);
      v16h fArl = load_afrag(parl, k0, sel);
      v16h fAih = load_afrag(paih, k0, sel);
      v16h fAil = load_afrag(pail, k0, sel);
      #pragma unroll
      for (int bt = 0; bt < 2; ++bt) {
        const int roff = (bt * 16 + col) * GP + k0 + sel * 16;
        v16h rh = *(const v16h*)(sh_rh + roff);
        v16h rl = *(const v16h*)(sh_rl + roff);
        // 3-term split-f16 ~ fp32 product
        tr[bt] = wmma16(fArh, rh, tr[bt]);
        tr[bt] = wmma16(fArh, rl, tr[bt]);
        tr[bt] = wmma16(fArl, rh, tr[bt]);
        ti[bt] = wmma16(fAih, rh, ti[bt]);
        ti[bt] = wmma16(fAih, rl, ti[bt]);
        ti[bt] = wmma16(fAil, rh, ti[bt]);
      }
    }
    __syncthreads();                  // all waves done reading sh_r
    #pragma unroll
    for (int bt = 0; bt < 2; ++bt) {
      int bl = bt * 16 + col;
      int b  = b0 + bl;
      #pragma unroll
      for (int j = 0; j < 8; ++j) {
        int g = g0 + j + 8 * sel;
        float sr = r[bt][j] - ga * tr[bt][j] + ga * dhyr[bt][j];
        float si = ga * (dhyi[bt][j] - ti[bt][j]);
        float nv = sqrtf(sr * sr + si * si) - th;   // relu(|s| - theta)
        nv = nv > 0.0f ? nv : 0.0f;
        r[bt][j] = nv;
        _Float16 h = (_Float16)nv;
        sh_rh[bl * GP + g] = h;
        sh_rl[bl * GP + g] = (_Float16)(nv - (float)h);
        if (g < G_TOT)
          out[OUT2 + (size_t)b * (S_TOT * L_TOT * G_TOT)
                   + (size_t)s * (L_TOT * G_TOT)
                   + (size_t)l * G_TOT + g] = nv;
      }
    }
    __syncthreads();                  // new sh_r visible to all waves
  }

  // ---- softmax over G for each batch column (cross-wave LDS reduction) ----
  const int bb   = tid & 31;          // batch column 0..31
  const int ch   = tid >> 5;          // g-chunk 0..11
  const int gbeg = ch * 16;
  const int gend = (gbeg + 16 < G_TOT) ? (gbeg + 16) : G_TOT;
  float mx = -3.4e38f;
  for (int g = gbeg; g < gend; ++g) {
    float v = (float)sh_rh[bb * GP + g] + (float)sh_rl[bb * GP + g];
    mx = fmaxf(mx, v);
  }
  sh_red[ch * 32 + bb] = mx;
  __syncthreads();
  if (tid < 32) {
    float m2 = -3.4e38f;
    for (int c = 0; c < 12; ++c) m2 = fmaxf(m2, sh_red[c * 32 + tid]);
    sh_stat[tid] = m2;
  }
  __syncthreads();
  const float bmax = sh_stat[bb];
  float sm = 0.0f;
  for (int g = gbeg; g < gend; ++g) {
    float v = (float)sh_rh[bb * GP + g] + (float)sh_rl[bb * GP + g];
    sm += __expf(v - bmax);
  }
  sh_red[ch * 32 + bb] = sm;
  __syncthreads();
  if (tid < 32) {
    float s2 = 0.0f;
    for (int c = 0; c < 12; ++c) s2 += sh_red[c * 32 + tid];
    sh_stat[tid] = s2;
  }
  __syncthreads();
  const float inv = 1.0f / sh_stat[bb];
  for (int g = gbeg; g < gend; ++g) {
    float v = (float)sh_rh[bb * GP + g] + (float)sh_rl[bb * GP + g];
    out[(size_t)(b0 + bb) * (S_TOT * G_TOT) + (size_t)s * G_TOT + g] = __expf(v - bmax) * inv;
  }
}

// ---------------------------- host launcher --------------------------------
extern "C" void kernel_launch(void* const* d_in, const int* in_sizes, int n_in,
                              void* d_out, int out_size, void* d_ws, size_t ws_size,
                              hipStream_t stream) {
  const float* covr  = (const float*)d_in[0];
  const float* covi  = (const float*)d_in[1];
  const float* dr    = (const float*)d_in[2];
  const float* di    = (const float*)d_in[3];
  const float* theta = (const float*)d_in[4];
  const float* gamma = (const float*)d_in[5];
  float* out = (float*)d_out;

  // workspace layout (split-f16 dictionaries + Gram matrices): ~3.4 MB total
  const size_t N_DT = (size_t)S_TOT * GP * M2;   // 110592
  const size_t N_A  = (size_t)S_TOT * GP * GP;   // 331776
  _Float16* w     = (_Float16*)d_ws;
  _Float16* drt_h = w;               _Float16* drt_l = drt_h + N_DT;
  _Float16* dit_h = drt_l + N_DT;    _Float16* dit_l = dit_h + N_DT;
  _Float16* arh   = dit_l + N_DT;    _Float16* arl   = arh + N_A;
  _Float16* aih   = arl + N_A;       _Float16* ail   = aih + N_A;

  prep_dict<<<(int)((N_DT + 255) / 256), 256, 0, stream>>>(dr, di, drt_h, drt_l, dit_h, dit_l);
  prep_gram<<<(int)((N_A + 255) / 256), 256, 0, stream>>>(dr, di, arh, arl, aih, ail);

  dim3 grid(B_TOT / BT, S_TOT);
  lista_main<<<grid, 384, 0, stream>>>(covr, covi, theta, gamma,
                                       drt_h, drt_l, dit_h, dit_l,
                                       arh, arl, aih, ail, out);
}
